// TransformerBlock_60464549593092
// MI455X (gfx1250) — matrix operand order repaired and checked by reading, not compiled, not run on hardware
//
#include <hip/hip_runtime.h>

// ---------------------------------------------------------------------------
// Types for CDNA5 WMMA (wave32): v_wmma_f32_16x16x32_bf16
// ---------------------------------------------------------------------------
typedef __attribute__((ext_vector_type(16))) __bf16 v16bf;
typedef __attribute__((ext_vector_type(8)))  __bf16 bf16x8;
typedef __attribute__((ext_vector_type(8)))  float  v8f;
typedef __attribute__((ext_vector_type(4)))  unsigned int u32x4;
typedef __attribute__((ext_vector_type(8)))  int i32x8;
typedef __attribute__((ext_vector_type(4)))  int i32x4;

#if defined(__has_builtin)
#if __has_builtin(__builtin_amdgcn_tensor_load_to_lds)
#define USE_TDM 1
#endif
#endif

__device__ __forceinline__ v8f wmma_bf16(v16bf a, v16bf b, v8f c) {
  // (neg_a, A, neg_b, B, c_mod, C, reuse_a, reuse_b)
  return __builtin_amdgcn_wmma_f32_16x16x32_bf16(false, a, false, b, (short)0, c,
                                                 false, false);
}
__device__ __forceinline__ v16bf pack16(bf16x8 lo, bf16x8 hi) {
  return __builtin_shufflevector(lo, hi, 0, 1, 2, 3, 4, 5, 6, 7,
                                 8, 9, 10, 11, 12, 13, 14, 15);
}

__device__ __forceinline__ void wait_tensorcnt0() {
#if defined(__has_builtin)
#if __has_builtin(__builtin_amdgcn_s_wait_tensorcnt)
  __builtin_amdgcn_s_wait_tensorcnt(0);
#else
  asm volatile("s_wait_tensorcnt 0x0" ::: "memory");
#endif
#else
  asm volatile("s_wait_tensorcnt 0x0" ::: "memory");
#endif
}

// Per-lane async copy: 16 bytes global -> LDS, tracked by ASYNCcnt.
__device__ __forceinline__ void async_copy_b128(unsigned lds_addr,
                                                unsigned long long gaddr) {
  asm volatile("global_load_async_to_lds_b128 %0, %1, off"
               :: "v"(lds_addr), "v"(gaddr)
               : "memory");
}
__device__ __forceinline__ void wait_asynccnt0() {
  asm volatile("s_wait_asynccnt 0x0" ::: "memory");
}

#ifdef USE_TDM
// Issue a TDM 2D tile load: tile (tile_w elems x tile_h rows) of 2-byte data,
// source rows strided by row_stride elems, into LDS at lds_addr.
// 6-arg builtin form: (u32x4 g0, i32x8 g1, i32x4 g2, i32x4 g3, i32x8, i32 cpol)
__device__ __forceinline__ void tdm_load_2d(unsigned lds_addr, const void* gptr,
                                            unsigned tile_w, unsigned tile_h,
                                            unsigned tensor_w, unsigned tensor_h,
                                            unsigned row_stride) {
  const unsigned long long ga = (unsigned long long)(__SIZE_TYPE__)gptr;
  u32x4 g0;
  g0[0] = 1u;                                    // count=1, user mode
  g0[1] = lds_addr;                              // lds_addr [63:32]
  g0[2] = (unsigned)(ga & 0xffffffffu);          // global_addr low
  g0[3] = (unsigned)((ga >> 32) & 0x01ffffffu) | (2u << 30);  // addr hi | type=2
  i32x8 g1;
  g1[0] = (int)(1u << 16);                       // wg_mask=0, data_size=1 (2B)
  g1[1] = (int)((tensor_w & 0xffffu) << 16);     // tensor_dim0[15:0]
  g1[2] = (int)(((tensor_w >> 16) & 0xffffu) | ((tensor_h & 0xffffu) << 16));
  g1[3] = (int)(((tensor_h >> 16) & 0xffffu) | ((tile_w & 0xffffu) << 16));
  g1[4] = (int)(tile_h & 0xffffu);               // tile_dim1 (tile_dim2=0)
  g1[5] = (int)row_stride;                       // tensor_dim0_stride[31:0]
  g1[6] = 0;                                     // stride0[47:32] | stride1 lo
  g1[7] = 0;
  i32x4 gz4 = {0, 0, 0, 0};
  i32x8 gz8 = {0, 0, 0, 0, 0, 0, 0, 0};
  __builtin_amdgcn_tensor_load_to_lds(g0, g1, gz4, gz4, gz8, 0);
}
#endif

// ---------------------------------------------------------------------------
// fp32 -> bf16 convert (grid-stride)
// ---------------------------------------------------------------------------
__global__ __launch_bounds__(256) void f32_to_bf16_kernel(
    const float* __restrict__ x, __bf16* __restrict__ y, long long n) {
  long long i = (long long)blockIdx.x * blockDim.x + threadIdx.x;
  long long stride = (long long)gridDim.x * blockDim.x;
  for (; i < n; i += stride) y[i] = (__bf16)x[i];
}

// ---------------------------------------------------------------------------
// RMSNorm over rows of [T, D] -> bf16
// ---------------------------------------------------------------------------
__global__ __launch_bounds__(256) void rmsnorm_bf16_kernel(
    const float* __restrict__ x, const float* __restrict__ w,
    __bf16* __restrict__ y, int D) {
  const int row = blockIdx.x;
  const float* xr = x + (size_t)row * D;
  float s = 0.f;
  for (int i = threadIdx.x; i < D; i += 256) { float v = xr[i]; s += v * v; }
  __shared__ float red[256];
  red[threadIdx.x] = s;
  __syncthreads();
  for (int st = 128; st > 0; st >>= 1) {
    if (threadIdx.x < st) red[threadIdx.x] += red[threadIdx.x + st];
    __syncthreads();
  }
  float n = rsqrtf(red[0] / (float)D + 1e-6f);
  for (int i = threadIdx.x; i < D; i += 256)
    y[(size_t)row * D + i] = (__bf16)(xr[i] * n * w[i]);
}

// ---------------------------------------------------------------------------
// Generic NT GEMM, bf16 inputs, f32 out:  C[M,N] (=|+=) rs*(A[M,K] @ B[N,K]^T) + resid
// Block: 256 threads (8 waves), tile 128x128x32. Wave grid 4(M) x 2(N),
// each wave: 2x4 grid of 16x16 WMMA tiles. Operand staging via TDM.
// ---------------------------------------------------------------------------
__global__ __launch_bounds__(256) void gemm_nt_bf16_kernel(
    const __bf16* __restrict__ A, const __bf16* __restrict__ Bw,
    float* __restrict__ C, int M, int N, int K,
    const float* __restrict__ resid, const float* __restrict__ rowscale,
    int rs_stride, int accumulate) {
  __shared__ __align__(16) __bf16 As[128 * 32];
  __shared__ __align__(16) __bf16 Bs[128 * 32];

  const int tid  = threadIdx.x;
  const int lane = tid & 31;
  const int wid  = tid >> 5;
  const int wm   = wid & 3;        // wave row (4 along M)
  const int wn   = wid >> 2;       // wave col (2 along N)
  const int lm   = lane & 15;
  const int hi   = lane >> 4;
  const int m0   = blockIdx.y * 128;
  const int n0   = blockIdx.x * 128;

  v8f zf = {};
  v8f acc[2][4];
#pragma unroll
  for (int a2 = 0; a2 < 2; ++a2)
#pragma unroll
    for (int c2 = 0; c2 < 4; ++c2) acc[a2][c2] = zf;

  const int ldr = tid >> 1;        // 0..127 (row within tile)
  const int ldp = (tid & 1) * 16;  // 0 or 16 (k offset)
  (void)ldr; (void)ldp;

  for (int k0 = 0; k0 < K; k0 += 32) {
#ifdef USE_TDM
    // DMA both operand tiles via the Tensor Data Mover; one issuing thread,
    // drain TENSORcnt, then publish to the workgroup with a barrier.
    if (tid == 0) {
      tdm_load_2d((unsigned)(__SIZE_TYPE__)&As[0], A + (size_t)m0 * K + k0,
                  32, 128, (unsigned)K, (unsigned)M, (unsigned)K);
      tdm_load_2d((unsigned)(__SIZE_TYPE__)&Bs[0], Bw + (size_t)n0 * K + k0,
                  32, 128, (unsigned)K, (unsigned)N, (unsigned)K);
      wait_tensorcnt0();
    }
    __syncthreads();
#else
    const __bf16* ga = A + (size_t)(m0 + ldr) * K + k0 + ldp;
    const __bf16* gb = Bw + (size_t)(n0 + ldr) * K + k0 + ldp;
    *(uint4*)&As[ldr * 32 + ldp]     = *(const uint4*)ga;
    *(uint4*)&As[ldr * 32 + ldp + 8] = *(const uint4*)(ga + 8);
    *(uint4*)&Bs[ldr * 32 + ldp]     = *(const uint4*)gb;
    *(uint4*)&Bs[ldr * 32 + ldp + 8] = *(const uint4*)(gb + 8);
    if (k0 + 32 < K) {
      __builtin_prefetch(ga + 32, 0, 1);   // global_prefetch_b8
      __builtin_prefetch(gb + 32, 0, 1);
    }
    __syncthreads();
#endif

    // B fragments (K x N): lane n = lm, element i -> K = hi*16 + i
    v16bf bfrag[4];
#pragma unroll
    for (int ni = 0; ni < 4; ++ni) {
      const __bf16* bp = Bs + (size_t)(wn * 64 + ni * 16 + lm) * 32 + hi * 16;
      bfrag[ni] = pack16(*(const bf16x8*)bp, *(const bf16x8*)(bp + 8));
    }
#pragma unroll
    for (int mi = 0; mi < 2; ++mi) {
      // A fragment: lane row m = lm; elem i -> K = hi*16 + i
      const __bf16* ap = As + (size_t)(wm * 32 + mi * 16 + lm) * 32;
      v16bf afrag = pack16(*(const bf16x8*)(ap + hi * 16),
                           *(const bf16x8*)(ap + hi * 16 + 8));
#pragma unroll
      for (int ni = 0; ni < 4; ++ni)
        acc[mi][ni] = wmma_bf16(afrag, bfrag[ni], acc[mi][ni]);
    }
    __syncthreads();
  }

  // Epilogue. C layout: VGPR i, lanes 0-15 -> M=i, lanes 16-31 -> M=i+8; N=lane%16.
#pragma unroll
  for (int mi = 0; mi < 2; ++mi) {
#pragma unroll
    for (int ni = 0; ni < 4; ++ni) {
      const int nn = n0 + wn * 64 + ni * 16 + lm;
#pragma unroll
      for (int i = 0; i < 8; ++i) {
        const int mm = m0 + wm * 32 + mi * 16 + i + 8 * hi;
        const size_t idx = (size_t)mm * N + nn;
        float v = acc[mi][ni][i];
        if (rowscale) v *= rowscale[(size_t)mm * rs_stride];
        if (resid) v += resid[idx];
        if (accumulate) C[idx] += v;
        else C[idx] = v;
      }
    }
  }
}

// ---------------------------------------------------------------------------
// Per-head RMSNorm + RoPE. x: [T, H*128] f32, out: [B,H,S,128] bf16.
// One wave per (token, head). hd = 128 fixed, half = 64.
// ---------------------------------------------------------------------------
__global__ __launch_bounds__(128) void headnorm_rope_kernel(
    const float* __restrict__ x, const float* __restrict__ w,
    __bf16* __restrict__ out, int S, int H) {
  const int wid = threadIdx.x >> 5, lane = threadIdx.x & 31;
  const long long idx = (long long)blockIdx.x * 4 + wid;  // over T*H
  const int t = (int)(idx / H), h = (int)(idx % H);
  const int b = t / S, s = t % S;
  const float* xr = x + (size_t)t * (H * 128) + h * 128;
  float v0 = xr[lane], v1 = xr[lane + 32], v2 = xr[lane + 64], v3 = xr[lane + 96];
  float ss = v0 * v0 + v1 * v1 + v2 * v2 + v3 * v3;
#pragma unroll
  for (int m = 1; m < 32; m <<= 1) ss += __shfl_xor(ss, m, 32);
  const float n = rsqrtf(ss * (1.f / 128.f) + 1e-6f);
  const float y0 = v0 * n * w[lane],      y1 = v1 * n * w[lane + 32];
  const float y2 = v2 * n * w[lane + 64], y3 = v3 * n * w[lane + 96];
  // freqs[j] = theta^(-j/64), pairs (j, j+64)
  const float kLogTheta = 13.815510558f;  // ln(1e6)
  const float f0 = __expf(-((float)lane) * (kLogTheta / 64.f));
  const float f1 = __expf(-((float)(lane + 32)) * (kLogTheta / 64.f));
  const float a0 = (float)s * f0, a1 = (float)s * f1;
  const float c0 = __cosf(a0), s0 = __sinf(a0);
  const float c1 = __cosf(a1), s1 = __sinf(a1);
  __bf16* orow = out + (((size_t)(b * H + h) * S) + s) * 128;
  orow[lane]      = (__bf16)(y0 * c0 - y2 * s0);
  orow[lane + 64] = (__bf16)(y2 * c0 + y0 * s0);
  orow[lane + 32] = (__bf16)(y1 * c1 - y3 * s1);
  orow[lane + 96] = (__bf16)(y3 * c1 + y1 * s1);
}

// v: [T, Hkv*128] f32 -> [B,Hkv,S,128] bf16
__global__ __launch_bounds__(256) void v_pack_kernel(
    const float* __restrict__ v, __bf16* __restrict__ out, int S, int Hkv,
    long long n) {
  long long i = (long long)blockIdx.x * blockDim.x + threadIdx.x;
  long long stride = (long long)gridDim.x * blockDim.x;
  for (; i < n; i += stride) {
    int d = (int)(i & 127);
    int h = (int)((i >> 7) % Hkv);
    int t = (int)(i / (128LL * Hkv));
    int b = t / S, s = t % S;
    out[(((size_t)(b * Hkv + h) * S) + s) * 128 + d] = (__bf16)v[i];
  }
}

// ---------------------------------------------------------------------------
// Flash attention. Q: [B,Hq,S,128] bf16, K/V: [B,Hkv,S,128] bf16,
// O: [B,S,Hq*128] bf16. Block = 4 waves x 16 q-rows = 64 q rows; 32-key chunks.
// K/V staged with global_load_async_to_lds_b128 (ASYNCcnt). grid = (S/64, Hq, B)
// ---------------------------------------------------------------------------
__global__ __launch_bounds__(128) void attn_kernel(
    const __bf16* __restrict__ Q, const __bf16* __restrict__ Kt,
    const __bf16* __restrict__ Vt, __bf16* __restrict__ O,
    int B, int S, int Hq, int Hkv) {
  __shared__ __align__(16) __bf16 Ks[32 * 128];
  __shared__ __align__(16) __bf16 Vs[32 * 128];
  __shared__ __align__(16) __bf16 Ps[4 * 16 * 32];

  const int wid = threadIdx.x >> 5;
  const int lane = threadIdx.x & 31;
  const int lm = lane & 15;
  const int hi = lane >> 4;
  const int b = blockIdx.z;
  const int h = blockIdx.y;
  const int hkv = h / (Hq / Hkv);
  const int q0blk = blockIdx.x * 64;
  const int qw0 = q0blk + wid * 16;

  // Q A-fragments for the wave's 16 rows, 4 chunks of K=32 over hd=128
  const __bf16* qrow = Q + (((size_t)(b * Hq + h) * S) + qw0 + lm) * 128;
  v16bf qa[4];
#pragma unroll
  for (int c = 0; c < 4; ++c)
    qa[c] = pack16(*(const bf16x8*)(qrow + c * 32 + hi * 16),
                   *(const bf16x8*)(qrow + c * 32 + hi * 16 + 8));

  v8f zf = {};
  v8f oacc[8];
  float mrun[8], lrun[8];
#pragma unroll
  for (int j = 0; j < 8; ++j) oacc[j] = zf;
#pragma unroll
  for (int i = 0; i < 8; ++i) { mrun[i] = -3.0e38f; lrun[i] = 0.f; }

  const float scale = 0.08838834764831845f;  // 1/sqrt(128)
  const size_t kvbase = ((size_t)(b * Hkv + hkv) * S) * 128;
  const int kv_end = q0blk + 64;  // causal upper bound (<= S)
  __bf16* Pw = Ps + wid * (16 * 32);
  const unsigned ks_lds = (unsigned)(__SIZE_TYPE__)&Ks[0];
  const unsigned vs_lds = (unsigned)(__SIZE_TYPE__)&Vs[0];

  for (int kk = 0; kk < kv_end; kk += 32) {
    // Stage K,V chunk [32 x 128] bf16 = 8KB each, async global->LDS (no VGPR trip)
    {
      const unsigned long long gk =
          (unsigned long long)(__SIZE_TYPE__)(Kt + kvbase + (size_t)kk * 128);
      const unsigned long long gv =
          (unsigned long long)(__SIZE_TYPE__)(Vt + kvbase + (size_t)kk * 128);
#pragma unroll
      for (int i = 0; i < 4; ++i) {
        const unsigned off = (unsigned)(threadIdx.x + i * 128) * 16u;
        async_copy_b128(ks_lds + off, gk + off);
        async_copy_b128(vs_lds + off, gv + off);
      }
      wait_asynccnt0();
    }
    __syncthreads();

    // Scores: 16 q-rows x 32 keys = two 16x16 WMMA accumulators
    v8f s0 = zf, s1 = zf;
#pragma unroll
    for (int c = 0; c < 4; ++c) {
      const __bf16* k0p = Ks + (size_t)lm * 128 + c * 32 + hi * 16;
      const __bf16* k1p = Ks + (size_t)(lm + 16) * 128 + c * 32 + hi * 16;
      v16bf bk0 = pack16(*(const bf16x8*)k0p, *(const bf16x8*)(k0p + 8));
      v16bf bk1 = pack16(*(const bf16x8*)k1p, *(const bf16x8*)(k1p + 8));
      s0 = wmma_bf16(qa[c], bk0, s0);
      s1 = wmma_bf16(qa[c], bk1, s1);
    }

    // Causal mask + online softmax (row reductions within 16-lane halves)
#pragma unroll
    for (int i = 0; i < 8; ++i) {
      const int qg = qw0 + i + 8 * hi;
      const int kg0 = kk + lm, kg1 = kk + 16 + lm;
      float a = s0[i] * scale, c = s1[i] * scale;
      a = (kg0 <= qg) ? a : -3.0e38f;
      c = (kg1 <= qg) ? c : -3.0e38f;
      float rm = fmaxf(a, c);
#pragma unroll
      for (int m = 1; m < 16; m <<= 1) rm = fmaxf(rm, __shfl_xor(rm, m, 32));
      const float mnew = fmaxf(mrun[i], rm);
      const float p0 = __expf(a - mnew), p1 = __expf(c - mnew);
      float rs = p0 + p1;
#pragma unroll
      for (int m = 1; m < 16; m <<= 1) rs += __shfl_xor(rs, m, 32);
      const float alpha = __expf(mrun[i] - mnew);
      lrun[i] = lrun[i] * alpha + rs;
      mrun[i] = mnew;
#pragma unroll
      for (int j = 0; j < 8; ++j) oacc[j][i] *= alpha;
      Pw[(i + 8 * hi) * 32 + lm] = (__bf16)p0;
      Pw[(i + 8 * hi) * 32 + 16 + lm] = (__bf16)p1;
    }
    __syncthreads();

    // P @ V  (P: 16x32 A-fragment from LDS; V B-fragments per 16-wide hd tile)
    {
      const __bf16* prow = Pw + (size_t)lm * 32;
      v16bf pa = pack16(*(const bf16x8*)(prow + hi * 16),
                        *(const bf16x8*)(prow + hi * 16 + 8));
#pragma unroll
      for (int j = 0; j < 8; ++j) {
        v16bf bv;
#pragma unroll
        for (int i2 = 0; i2 < 16; ++i2)
          bv[i2] = Vs[(size_t)(hi * 16 + i2) * 128 + j * 16 + lm];
        oacc[j] = wmma_bf16(pa, bv, oacc[j]);
      }
    }
    __syncthreads();
  }

  // Write O as [B,S,Hq,128] bf16
#pragma unroll
  for (int i = 0; i < 8; ++i) {
    const int qg = qw0 + i + 8 * hi;
    const float inv = 1.f / lrun[i];
    const size_t orow = ((size_t)(b * S + qg) * Hq + h) * 128;
#pragma unroll
    for (int j = 0; j < 8; ++j)
      O[orow + j * 16 + lm] = (__bf16)(oacc[j][i] * inv);
  }
}

// ---------------------------------------------------------------------------
// Router: logits = tok @ gate_w^T, softmax, top-2, normalized combine weights.
// One wave per token.
// ---------------------------------------------------------------------------
__global__ __launch_bounds__(256) void router_kernel(
    const __bf16* __restrict__ tok, const float* __restrict__ gw,
    float* __restrict__ combine, int T, int D, int E) {
  const int wid = threadIdx.x >> 5, lane = threadIdx.x & 31;
  const int t = blockIdx.x * 8 + wid;
  if (t >= T) return;
  float logit[8];
  for (int e = 0; e < E; ++e) {
    const __bf16* tr = tok + (size_t)t * D;
    const float* wr = gw + (size_t)e * D;
    float s = 0.f;
    for (int k = lane; k < D; k += 32) s += (float)tr[k] * wr[k];
#pragma unroll
    for (int m = 1; m < 32; m <<= 1) s += __shfl_xor(s, m, 32);
    logit[e] = s;
  }
  if (lane == 0) {
    float mx = logit[0];
    for (int e = 1; e < E; ++e) mx = fmaxf(mx, logit[e]);
    float pe[8], sum = 0.f;
    for (int e = 0; e < E; ++e) { pe[e] = __expf(logit[e] - mx); sum += pe[e]; }
    const float inv = 1.f / sum;
    for (int e = 0; e < E; ++e) pe[e] *= inv;
    int i1 = 0;
    for (int e = 1; e < E; ++e) if (pe[e] > pe[i1]) i1 = e;
    int i2 = (i1 == 0) ? 1 : 0;
    for (int e = 0; e < E; ++e) if (e != i1 && pe[e] > pe[i2]) i2 = e;
    const float tot = pe[i1] + pe[i2] + 1e-9f;
    for (int e = 0; e < E; ++e) combine[(size_t)t * E + e] = 0.f;
    combine[(size_t)t * E + i1] = pe[i1] / tot;
    combine[(size_t)t * E + i2] = pe[i2] / tot;
  }
}

// act = silu(eg) * eu -> bf16
__global__ __launch_bounds__(256) void moe_act_kernel(
    const float* __restrict__ eg, const float* __restrict__ eu,
    __bf16* __restrict__ act, long long n) {
  long long i = (long long)blockIdx.x * blockDim.x + threadIdx.x;
  long long stride = (long long)gridDim.x * blockDim.x;
  for (; i < n; i += stride) {
    const float g = eg[i];
    const float s = g / (1.f + __expf(-g));
    act[i] = (__bf16)(s * eu[i]);
  }
}

// ---------------------------------------------------------------------------
// Host orchestration
// ---------------------------------------------------------------------------
extern "C" void kernel_launch(void* const* d_in, const int* in_sizes, int n_in,
                              void* d_out, int out_size, void* d_ws,
                              size_t ws_size, hipStream_t stream) {
  (void)in_sizes; (void)n_in; (void)out_size; (void)ws_size;
  // Problem dims (fixed by the reference's setup_inputs)
  const int B = 2, S = 2048, D = 2048, Hq = 16, Hkv = 4, hd = 128;
  const int E = 8, I = 1408;
  const int T = B * S;                 // 4096
  const int Dq = Hq * hd;              // 2048
  const int Dkv = Hkv * hd;            // 512

  const float* hidden      = (const float*)d_in[0];
  const float* attn_norm_w = (const float*)d_in[1];
  const float* q_w         = (const float*)d_in[2];
  const float* k_w         = (const float*)d_in[3];
  const float* v_w         = (const float*)d_in[4];
  const float* o_w         = (const float*)d_in[5];
  const float* q_norm_w    = (const float*)d_in[6];
  const float* k_norm_w    = (const float*)d_in[7];
  const float* ffn_norm_w  = (const float*)d_in[8];
  const float* gate_w      = (const float*)d_in[9];
  const float* wg          = (const float*)d_in[10];
  const float* wu          = (const float*)d_in[11];
  const float* wd          = (const float*)d_in[12];
  float* out = (float*)d_out;

  // Workspace carve-up
  char* ws = (char*)d_ws;
  auto alloc = [&](size_t bytes) -> char* {
    char* p = ws;
    ws += (bytes + 255) & ~(size_t)255;
    return p;
  };
  __bf16* qw_bf  = (__bf16*)alloc((size_t)Dq * D * 2);
  __bf16* kw_bf  = (__bf16*)alloc((size_t)Dkv * D * 2);
  __bf16* vw_bf  = (__bf16*)alloc((size_t)Dkv * D * 2);
  __bf16* ow_bf  = (__bf16*)alloc((size_t)D * Dq * 2);
  __bf16* wg_bf  = (__bf16*)alloc((size_t)E * I * D * 2);
  __bf16* wu_bf  = (__bf16*)alloc((size_t)E * I * D * 2);
  __bf16* wd_bf  = (__bf16*)alloc((size_t)E * D * I * 2);
  __bf16* h_bf   = (__bf16*)alloc((size_t)T * D * 2);
  float*  q_f    = (float*)alloc((size_t)T * Dq * 4);
  float*  k_f    = (float*)alloc((size_t)T * Dkv * 4);
  float*  v_f    = (float*)alloc((size_t)T * Dkv * 4);
  __bf16* q_r    = (__bf16*)alloc((size_t)T * Dq * 2);
  __bf16* k_r    = (__bf16*)alloc((size_t)T * Dkv * 2);
  __bf16* v_r    = (__bf16*)alloc((size_t)T * Dkv * 2);
  __bf16* attn_bf= (__bf16*)alloc((size_t)T * Dq * 2);
  __bf16* tok_bf = (__bf16*)alloc((size_t)T * D * 2);
  float*  eg_f   = (float*)alloc((size_t)T * I * 4);
  float*  eu_f   = (float*)alloc((size_t)T * I * 4);
  __bf16* act_bf = (__bf16*)alloc((size_t)T * I * 2);
  float*  combine= (float*)alloc((size_t)T * E * 4);

  // --- weight downconvert to bf16 ---
  f32_to_bf16_kernel<<<4096, 256, 0, stream>>>(q_w, qw_bf, (long long)Dq * D);
  f32_to_bf16_kernel<<<4096, 256, 0, stream>>>(k_w, kw_bf, (long long)Dkv * D);
  f32_to_bf16_kernel<<<4096, 256, 0, stream>>>(v_w, vw_bf, (long long)Dkv * D);
  f32_to_bf16_kernel<<<4096, 256, 0, stream>>>(o_w, ow_bf, (long long)D * Dq);
  f32_to_bf16_kernel<<<8192, 256, 0, stream>>>(wg, wg_bf, (long long)E * I * D);
  f32_to_bf16_kernel<<<8192, 256, 0, stream>>>(wu, wu_bf, (long long)E * I * D);
  f32_to_bf16_kernel<<<8192, 256, 0, stream>>>(wd, wd_bf, (long long)E * D * I);

  // --- attention input norm ---
  rmsnorm_bf16_kernel<<<T, 256, 0, stream>>>(hidden, attn_norm_w, h_bf, D);

  // --- QKV projections (WMMA GEMMs) ---
  gemm_nt_bf16_kernel<<<dim3(Dq / 128, T / 128), 256, 0, stream>>>(
      h_bf, qw_bf, q_f, T, Dq, D, nullptr, nullptr, 0, 0);
  gemm_nt_bf16_kernel<<<dim3(Dkv / 128, T / 128), 256, 0, stream>>>(
      h_bf, kw_bf, k_f, T, Dkv, D, nullptr, nullptr, 0, 0);
  gemm_nt_bf16_kernel<<<dim3(Dkv / 128, T / 128), 256, 0, stream>>>(
      h_bf, vw_bf, v_f, T, Dkv, D, nullptr, nullptr, 0, 0);

  // --- per-head RMSNorm + RoPE, V relayout ---
  headnorm_rope_kernel<<<(T * Hq) / 4, 128, 0, stream>>>(q_f, q_norm_w, q_r, S, Hq);
  headnorm_rope_kernel<<<(T * Hkv) / 4, 128, 0, stream>>>(k_f, k_norm_w, k_r, S, Hkv);
  v_pack_kernel<<<4096, 256, 0, stream>>>(v_f, v_r, S, Hkv, (long long)T * Dkv);

  // --- flash attention (WMMA + async LDS staging) ---
  attn_kernel<<<dim3(S / 64, Hq, B), 128, 0, stream>>>(q_r, k_r, v_r, attn_bf,
                                                       B, S, Hq, Hkv);

  // --- O projection + residual -> d_out (f32) ---
  gemm_nt_bf16_kernel<<<dim3(D / 128, T / 128), 256, 0, stream>>>(
      attn_bf, ow_bf, out, T, D, Dq, hidden, nullptr, 0, 0);

  // --- FFN norm + router ---
  rmsnorm_bf16_kernel<<<T, 256, 0, stream>>>(out, ffn_norm_w, tok_bf, D);
  router_kernel<<<T / 8, 256, 0, stream>>>(tok_bf, gate_w, combine, T, D, E);

  // --- MoE experts: gate/up GEMMs, SiLU*up, down GEMM scaled by combine ---
  for (int e = 0; e < E; ++e) {
    const __bf16* wge = wg_bf + (size_t)e * I * D;
    const __bf16* wue = wu_bf + (size_t)e * I * D;
    const __bf16* wde = wd_bf + (size_t)e * D * I;
    gemm_nt_bf16_kernel<<<dim3(I / 128, T / 128), 256, 0, stream>>>(
        tok_bf, wge, eg_f, T, I, D, nullptr, nullptr, 0, 0);
    gemm_nt_bf16_kernel<<<dim3(I / 128, T / 128), 256, 0, stream>>>(
        tok_bf, wue, eu_f, T, I, D, nullptr, nullptr, 0, 0);
    moe_act_kernel<<<8192, 256, 0, stream>>>(eg_f, eu_f, act_bf, (long long)T * I);
    // out[t,:] += combine[t,e] * (act @ wd[e]^T)
    gemm_nt_bf16_kernel<<<dim3(D / 128, T / 128), 256, 0, stream>>>(
        act_bf, wde, out, T, D, I, nullptr, combine + e, E, 1);
  }
}
